// Head_67903432950060
// MI455X (gfx1250) — matrix operand order repaired and checked by reading, not compiled, not run on hardware
//
#include <hip/hip_runtime.h>
#include <hip/hip_bf16.h>

typedef _Float16 v16h __attribute__((ext_vector_type(16)));
typedef _Float16 v8h  __attribute__((ext_vector_type(8)));
typedef float    v8f  __attribute__((ext_vector_type(8)));
typedef unsigned int u32x4 __attribute__((ext_vector_type(4)));
typedef int          i32x4 __attribute__((ext_vector_type(4)));
typedef int          i32x8 __attribute__((ext_vector_type(8)));

#define BB 16
#define SS 2048
#define DM 384
#define DQ 64

#if defined(__has_builtin)
#if __has_builtin(__builtin_amdgcn_tensor_load_to_lds) && \
    __has_builtin(__builtin_amdgcn_s_wait_tensorcnt)
#define HAVE_TDM 1
#endif
#endif

static __device__ __forceinline__ v16h cat8(v8h lo, v8h hi) {
    return __builtin_shufflevector(lo, hi, 0,1,2,3,4,5,6,7,8,9,10,11,12,13,14,15);
}

static __device__ __forceinline__ v8f wmma_f16(v16h a, v16h b, v8f c) {
    return __builtin_amdgcn_wmma_f32_16x16x32_f16(false, a, false, b, (short)0, c, false, false);
}

static __device__ __forceinline__ unsigned lds_offset(const void* p) {
    // generic -> LDS addrspace cast; ptrtoint yields the 32-bit LDS byte offset
    return (unsigned)(size_t)(__attribute__((address_space(3))) const void*)p;
}

#ifdef HAVE_TDM
// Issue a 2D TDM tile load: tile_d0 x tile_d1 elements of 2 bytes,
// row stride `stride0` (elements), contiguous into LDS at `lds_addr`.
static __device__ __forceinline__ void tdm_load_2d(unsigned lds_addr,
                                                   unsigned long long gaddr,
                                                   unsigned tile_d0, unsigned tile_d1,
                                                   unsigned stride0) {
    // D# group0: count=1 | lds_addr | global_addr[56:0] | type=2 (bits 127:126)
    u32x4 g0 = { 1u, lds_addr, (unsigned)gaddr,
                 ((unsigned)(gaddr >> 32) & 0x01FFFFFFu) | 0x80000000u };
    // D# group1: data_size=2B (code 1) at bits[17:16]; tensor_dim0=tile_d0,
    // tensor_dim1=tile_d1 (exact-fit, no OOB); tile dims; dim0 stride.
    i32x8 g1 = { 0x00010000,
                 (int)(tile_d0 << 16),          // d1[31:16] = tensor_dim0[15:0]
                 (int)(tile_d1 << 16),          // d2[31:16] = tensor_dim1[15:0]
                 (int)(tile_d0 << 16),          // d3[31:16] = tile_dim0
                 (int)tile_d1,                  // d4[15:0]  = tile_dim1
                 (int)stride0,                  // d5 = tensor_dim0_stride[31:0]
                 0, 0 };
    i32x4 z4 = { 0, 0, 0, 0 };
#if __has_include(<hip/amd_detail/amd_gfx1250_TDM.h>)
    i32x8 z8 = { 0, 0, 0, 0, 0, 0, 0, 0 };
    __builtin_amdgcn_tensor_load_to_lds(g0, g1, z4, z4, z8, 0);   // clang-23 form
#else
    __builtin_amdgcn_tensor_load_to_lds(g0, g1, z4, z4, 0);       // ROCm 7.2 form
#endif
}
#endif

// ---------------------------------------------------------------------------
// Kernel 1: Wq/Wk/Wv (f32 [384][64]) -> transposed f16 Wt[3][64][384];
// fold the 1/sqrt(d_q)=0.125 score scale into Wq.
// ---------------------------------------------------------------------------
__global__ __launch_bounds__(256)
void convw_kernel(const float* __restrict__ wq, const float* __restrict__ wk,
                  const float* __restrict__ wv, _Float16* __restrict__ wt) {
    int tid = blockIdx.x * 256 + threadIdx.x;
    const int per = DM * DQ;
    if (tid >= 3 * per) return;
    int mat = tid / per;
    int r   = tid % per;
    int d   = r / DQ;
    int n   = r % DQ;
    const float* w = (mat == 0) ? wq : (mat == 1) ? wk : wv;
    float scale = (mat == 0) ? 0.125f : 1.0f;
    wt[(size_t)(mat * DQ + n) * DM + d] = (_Float16)(w[d * DQ + n] * scale);
}

// ---------------------------------------------------------------------------
// Kernel 2: fused QKV projection.  One wave = 16 rows x 64 cols of Q, K, V.
// B tiles for a whole k-step are batch-loaded first so the global loads
// clause together and overlap the WMMA stream.  V is stored TRANSPOSED
// (Vt[B][64][S]) so the attention kernel's V tile is a plain 2D copy.
// ---------------------------------------------------------------------------
__global__ __launch_bounds__(128)
void qkv_proj_kernel(const float* __restrict__ x, const _Float16* __restrict__ wt,
                     _Float16* __restrict__ qh, _Float16* __restrict__ kh,
                     _Float16* __restrict__ vt) {
    const int wv_id = threadIdx.x >> 5;
    const int lane  = threadIdx.x & 31;
    const int m     = lane & 15;
    const int half  = lane >> 4;
    const int strip = blockIdx.x * 4 + wv_id;
    const int g0    = strip * 16;

    const float* xrow = x + (size_t)(g0 + m) * DM;

    v8f acc[12];
#pragma unroll
    for (int i = 0; i < 12; ++i) acc[i] = (v8f){0,0,0,0,0,0,0,0};

#pragma unroll
    for (int ks = 0; ks < 12; ++ks) {
        const int kk = ks * 32;
        v16h A;
        const float* pa = xrow + kk + half * 16;
        const float* pb = xrow + kk + half * 16 + 8;
#pragma unroll
        for (int q = 0; q < 8; ++q) A[q]     = (_Float16)pa[q];
#pragma unroll
        for (int q = 0; q < 8; ++q) A[8 + q] = (_Float16)pb[q];

        // batch all 12 B tiles first -> loads clause, waits stagger
        v16h Bt[12];
#pragma unroll
        for (int idx = 0; idx < 12; ++idx) {
            const int mat = idx >> 2, t = idx & 3;
            Bt[idx] = *(const v16h*)(wt +
                (size_t)(mat * DQ + t * 16 + m) * DM + kk + half * 16);
        }
#pragma unroll
        for (int idx = 0; idx < 12; ++idx)
            acc[idx] = wmma_f16(A, Bt[idx], acc[idx]);
    }

#pragma unroll
    for (int idx = 0; idx < 12; ++idx) {
        const int mat = idx >> 2, t = idx & 3;
#pragma unroll
        for (int i = 0; i < 8; ++i) {
            int row = g0 + i + 8 * half;            // global row in [0, B*S)
            float v = acc[idx][i];
            if (mat == 0) {
                qh[(size_t)row * DQ + t * 16 + m] = (_Float16)v;
            } else if (mat == 1) {
                kh[(size_t)row * DQ + t * 16 + m] = (_Float16)v;
            } else {
                int b = row >> 11, s = row & (SS - 1);
                vt[((size_t)b * DQ + t * 16 + m) * SS + s] = (_Float16)v;
            }
        }
    }
}

// ---------------------------------------------------------------------------
// Kernel 3: flash attention, TDM double-buffered K/V tiles.
// Block = 64 query rows (4 waves x 16); column tiles of 32 keys/values.
// ---------------------------------------------------------------------------
__global__ __launch_bounds__(128)
void flash_attn_kernel(const _Float16* __restrict__ qh, const _Float16* __restrict__ kh,
                       const _Float16* __restrict__ vt, float* __restrict__ out) {
    __shared__ __align__(32) _Float16 lds_k [2][32][64];   // [buf][col][d]
    __shared__ __align__(32) _Float16 lds_vt[2][64][32];   // [buf][d][col]
    __shared__ __align__(32) _Float16 lds_p [4][16][32];   // per-wave P scratch

    const int w    = threadIdx.x >> 5;
    const int lane = threadIdx.x & 31;
    const int m    = lane & 15;
    const int half = lane >> 4;

    const int b  = blockIdx.x / (SS / 64);
    const int rt = blockIdx.x % (SS / 64);
    const int r0 = rt * 64;
    const size_t bS = (size_t)b * SS;

    const _Float16* qrow = qh + (bS + r0 + w * 16 + m) * DQ;
    const v16h A0 = cat8(*(const v8h*)(qrow +      half * 16),
                         *(const v8h*)(qrow + 8  + half * 16));
    const v16h A1 = cat8(*(const v8h*)(qrow + 32 + half * 16),
                         *(const v8h*)(qrow + 40 + half * 16));

    v8f o[4];
#pragma unroll
    for (int t = 0; t < 4; ++t) o[t] = (v8f){0,0,0,0,0,0,0,0};
    float mi[8], li[8];
#pragma unroll
    for (int i = 0; i < 8; ++i) { mi[i] = -1e30f; li[i] = 0.0f; }

    const int rbase  = r0 + w * 16 + 8 * half;
    const int ntiles = r0 / 32 + 2;

#ifdef HAVE_TDM
    const unsigned long long kbase = (unsigned long long)(size_t)kh;
    const unsigned long long vbase = (unsigned long long)(size_t)vt;
    if (w == 0) {   // prologue: DMA tile 0 into buffer 0
        tdm_load_2d(lds_offset(&lds_k [0][0][0]), kbase + ((bS + 0) * DQ) * 2, 64, 32, 64);
        tdm_load_2d(lds_offset(&lds_vt[0][0][0]), vbase + (bS * DQ + 0) * 2,   32, 64, SS);
    }
#endif

    for (int ct = 0; ct < ntiles; ++ct) {
        const int c0  = ct * 32;
        const int buf = ct & 1;

#ifdef HAVE_TDM
        if (w == 0) {
            if (ct + 1 < ntiles) {
                const int c1 = c0 + 32, nb = buf ^ 1;
                tdm_load_2d(lds_offset(&lds_k [nb][0][0]),
                            kbase + ((bS + c1) * DQ) * 2, 64, 32, 64);
                tdm_load_2d(lds_offset(&lds_vt[nb][0][0]),
                            vbase + (bS * DQ + c1) * 2,   32, 64, SS);
                __builtin_amdgcn_s_wait_tensorcnt(2);   // current 2 DMAs done
            } else {
                __builtin_amdgcn_s_wait_tensorcnt(0);
            }
        }
#else
        for (int e = threadIdx.x; e < 32 * 64; e += 128) {
            int col = e >> 6, d = e & 63;
            lds_k[buf][col][d] = kh[(bS + c0 + col) * DQ + d];
        }
        for (int e = threadIdx.x; e < 64 * 32; e += 128) {
            int d = e >> 5, col = e & 31;
            lds_vt[buf][d][col] = vt[(bS * DQ) + (size_t)d * SS + c0 + col];
        }
#endif
        __syncthreads();   // tile `buf` ready for all waves

        // ---- scores S = Q K^T (16x32) ----
        v8f sc0 = (v8f){0,0,0,0,0,0,0,0};
        v8f sc1 = (v8f){0,0,0,0,0,0,0,0};
        {
            const v16h B00 = *(const v16h*)(&lds_k[buf][m][half * 16]);
            const v16h B01 = *(const v16h*)(&lds_k[buf][m][32 + half * 16]);
            const v16h B10 = *(const v16h*)(&lds_k[buf][16 + m][half * 16]);
            const v16h B11 = *(const v16h*)(&lds_k[buf][16 + m][32 + half * 16]);
            sc0 = wmma_f16(A0, B00, sc0);
            sc0 = wmma_f16(A1, B01, sc0);
            sc1 = wmma_f16(A0, B10, sc1);
            sc1 = wmma_f16(A1, B11, sc1);
        }

        // ---- causal mask ----
#pragma unroll
        for (int i = 0; i < 8; ++i) {
            int row = rbase + i;
            if (c0 + m      > row) sc0[i] = -1e30f;
            if (c0 + 16 + m > row) sc1[i] = -1e30f;
        }

        // ---- online softmax ----
#pragma unroll
        for (int i = 0; i < 8; ++i) {
            float rm = fmaxf(sc0[i], sc1[i]);
            rm = fmaxf(rm, __shfl_xor(rm, 1, 32));
            rm = fmaxf(rm, __shfl_xor(rm, 2, 32));
            rm = fmaxf(rm, __shfl_xor(rm, 4, 32));
            rm = fmaxf(rm, __shfl_xor(rm, 8, 32));
            float mn    = fmaxf(mi[i], rm);
            float alpha = __expf(mi[i] - mn);
            mi[i] = mn;
            float p0 = __expf(sc0[i] - mn);
            float p1 = __expf(sc1[i] - mn);
            float rs = p0 + p1;
            rs += __shfl_xor(rs, 1, 32);
            rs += __shfl_xor(rs, 2, 32);
            rs += __shfl_xor(rs, 4, 32);
            rs += __shfl_xor(rs, 8, 32);
            li[i] = li[i] * alpha + rs;
            o[0][i] *= alpha; o[1][i] *= alpha; o[2][i] *= alpha; o[3][i] *= alpha;
            lds_p[w][i + 8 * half][m]      = (_Float16)p0;
            lds_p[w][i + 8 * half][16 + m] = (_Float16)p1;
        }

        // ---- P as A tile (same-wave DS ops are in-order) ----
        const v16h Ap = cat8(*(const v8h*)(&lds_p[w][m][half * 16]),
                             *(const v8h*)(&lds_p[w][m][8 + half * 16]));

        // ---- O += P V ----
#pragma unroll
        for (int t = 0; t < 4; ++t) {
            const v16h Bv = *(const v16h*)(&lds_vt[buf][t * 16 + m][half * 16]);
            o[t] = wmma_f16(Ap, Bv, o[t]);
        }

        __syncthreads();   // reads of `buf` done before DMA overwrites it
    }

#pragma unroll
    for (int i = 0; i < 8; ++i) {
        float inv = 1.0f / li[i];
        int row = rbase + i;
#pragma unroll
        for (int t = 0; t < 4; ++t)
            out[(bS + row) * DQ + t * 16 + m] = o[t][i] * inv;
    }
}

// ---------------------------------------------------------------------------
extern "C" void kernel_launch(void* const* d_in, const int* in_sizes, int n_in,
                              void* d_out, int out_size, void* d_ws, size_t ws_size,
                              hipStream_t stream) {
    (void)in_sizes; (void)n_in; (void)out_size; (void)ws_size;
    const float* x  = (const float*)d_in[0];
    const float* wq = (const float*)d_in[1];
    const float* wk = (const float*)d_in[2];
    const float* wv = (const float*)d_in[3];
    float* out = (float*)d_out;

    const size_t nqkv = (size_t)BB * SS * DQ;
    _Float16* qh = (_Float16*)d_ws;
    _Float16* kh = qh + nqkv;
    _Float16* vt = kh + nqkv;          // stored transposed: [B][64][S]
    _Float16* wt = vt + nqkv;

    const int wtot = 3 * DM * DQ;
    convw_kernel<<<(wtot + 255) / 256, 256, 0, stream>>>(wq, wk, wv, wt);

    const int strips = (BB * SS) / 16;
    qkv_proj_kernel<<<strips / 4, 128, 0, stream>>>(x, wt, qh, kh, vt);

    const int ablocks = BB * (SS / 64);
    flash_attn_kernel<<<ablocks, 128, 0, stream>>>(qh, kh, vt, out);
}
